// GCN_72851235275095
// MI455X (gfx1250) — compile-verified, matrix operand order checked by reading, not run on hardware
//
#include <hip/hip_runtime.h>
#include <hip/hip_bf16.h>

// ---------------- types for WMMA ----------------
typedef __attribute__((ext_vector_type(16))) __bf16 v16bf;
typedef __attribute__((ext_vector_type(8)))  float  v8f;

union Frag16 {
    uint4  u[2];   // two 16B chunks = 16 bf16 elements
    v16bf  v;
};

#define N_NODES   50000
#define N_EDGES   1600000
#define HID       128
#define N_GRAPHS  512
#define N_CLASSES 10

// round-to-nearest-even fp32 -> bf16 (raw bits)
__device__ __forceinline__ unsigned short f2bf(float f) {
    unsigned int u = __float_as_uint(f);
    if ((u & 0x7F800000u) == 0x7F800000u)            // inf / nan: truncate
        return (unsigned short)(u >> 16);
    unsigned int r = u + 0x7FFFu + ((u >> 16) & 1u); // RNE
    return (unsigned short)(r >> 16);
}

// ---------------- small utility kernels ----------------
__global__ void zero_f32(float* __restrict__ p, int n) {
    int i = blockIdx.x * blockDim.x + threadIdx.x;
    if (i < n) p[i] = 0.0f;
}

__global__ void deg_count(const int* __restrict__ dst, float* __restrict__ deg, int E) {
    int e = blockIdx.x * blockDim.x + threadIdx.x;
    if (e < E) atomicAdd(&deg[dst[e]], 1.0f);
}

// deg -> deg_inv_sqrt in place (self-loop included: deg+1)
__global__ void finish_dis(float* __restrict__ deg, int n) {
    int i = blockIdx.x * blockDim.x + threadIdx.x;
    if (i < n) deg[i] = rsqrtf(deg[i] + 1.0f);
}

// fp32 features -> bf16 staging (optionally fused ReLU on the input)
__global__ void cvt_x(const float* __restrict__ f, unsigned short* __restrict__ xb,
                      int n, int relu) {
    int i = blockIdx.x * blockDim.x + threadIdx.x;
    if (i >= n) return;
    float v = f[i];
    if (relu) v = fmaxf(v, 0.0f);
    xb[i] = f2bf(v);
}

// W[k][n] (row-major 128x128) -> Wt bf16 [n][k]
__global__ void cvt_wt(const float* __restrict__ W, unsigned short* __restrict__ wt) {
    int i = blockIdx.x * blockDim.x + threadIdx.x;   // i = n*128 + k
    if (i >= HID * HID) return;
    int n = i >> 7, k = i & 127;
    wt[i] = f2bf(W[k * HID + n]);
}

// ---------------- WMMA GEMM: h[M][128] = Xb[M][128] @ W  ----------------
// Xb bf16 row-major [M][128]; Wt bf16 [n][k] (transposed W). One wave computes a
// 16x16 tile; block = 8 waves -> 16 rows x 128 cols; grid.x = M/16.
__global__ __launch_bounds__(256) void gemm_bf16_wmma(
    const unsigned short* __restrict__ xb,
    const unsigned short* __restrict__ wt,
    float* __restrict__ hout, int M)
{
    const int lane = threadIdx.x & 31;
    const int wave = threadIdx.x >> 5;      // 0..7 -> column tile
    const int r    = lane & 15;
    const int half = lane >> 4;
    const int m0   = blockIdx.x * 16;
    const int n0   = wave * 16;
    if (m0 >= M) return;

    v8f acc = {0.f, 0.f, 0.f, 0.f, 0.f, 0.f, 0.f, 0.f};

    const uint4* arow = reinterpret_cast<const uint4*>(xb + (size_t)(m0 + r) * HID);
    const uint4* brow = reinterpret_cast<const uint4*>(wt + (size_t)(n0 + r) * HID);

#pragma unroll
    for (int kt = 0; kt < 4; ++kt) {
        // chunk element offsets: k0 + half*8 and k0 + 16 + half*8  (uint4 = 8 elems)
        const int e0 = kt * 4 + half;
        Frag16 a, b;
        a.u[0] = arow[e0]; a.u[1] = arow[e0 + 2];
        b.u[0] = brow[e0]; b.u[1] = brow[e0 + 2];
        acc = __builtin_amdgcn_wmma_f32_16x16x32_bf16(
            /*neg_a=*/false, a.v, /*neg_b=*/false, b.v,
            /*c_mod=*/(short)0, acc, /*reuse_a=*/false, /*reuse_b=*/false);
    }

    // C/D layout: VGPR j -> row j + 8*half; col = n0 + r
    float* ocol = hout + (size_t)m0 * HID + n0 + r;
#pragma unroll
    for (int j = 0; j < 8; ++j) {
        int row = j + half * 8;
        ocol[(size_t)row * HID] = acc[j];
    }
}

// agg[i][c] = h[i][c] * dis[i]^2 + bias[c]   (self-loop + bias)
__global__ void selfloop_init(const float* __restrict__ h, const float* __restrict__ dis,
                              const float* __restrict__ bias, float* __restrict__ agg, int n) {
    int i = blockIdx.x * blockDim.x + threadIdx.x;   // over N*128
    if (i >= n) return;
    int node = i >> 7, c = i & 127;
    float d = dis[node];
    agg[i] = h[i] * d * d + bias[c];
}

// wave per edge: agg[dst] += h[src] * (dis[src]*dis[dst]); lane handles 4 cols
__global__ void scatter_edges(const int* __restrict__ src, const int* __restrict__ dst,
                              const float* __restrict__ dis, const float* __restrict__ h,
                              float* __restrict__ agg, int E)
{
    int wid  = (blockIdx.x * blockDim.x + threadIdx.x) >> 5;
    int lane = threadIdx.x & 31;
    if (wid >= E) return;
    int s = src[wid], d = dst[wid];
    float nrm = dis[s] * dis[d];
    float4 v = reinterpret_cast<const float4*>(h + (size_t)s * HID)[lane];
    float* ap = agg + (size_t)d * HID + lane * 4;
    atomicAdd(ap + 0, v.x * nrm);
    atomicAdd(ap + 1, v.y * nrm);
    atomicAdd(ap + 2, v.z * nrm);
    atomicAdd(ap + 3, v.w * nrm);
}

// wave per node: pooled[batch[i]] += h4[i]; counts[batch[i]] += 1
__global__ void pool_nodes(const float* __restrict__ h, const int* __restrict__ batch,
                           float* __restrict__ pooled, float* __restrict__ counts, int n)
{
    int wid  = (blockIdx.x * blockDim.x + threadIdx.x) >> 5;
    int lane = threadIdx.x & 31;
    if (wid >= n) return;
    int g = batch[wid];
    if (lane == 0) atomicAdd(&counts[g], 1.0f);
    float4 v = reinterpret_cast<const float4*>(h + (size_t)wid * HID)[lane];
    float* pp = pooled + (size_t)g * HID + lane * 4;
    atomicAdd(pp + 0, v.x);
    atomicAdd(pp + 1, v.y);
    atomicAdd(pp + 2, v.z);
    atomicAdd(pp + 3, v.w);
}

// one thread per graph: mean, logits (128x10), softmax
__global__ void head_kernel(const float* __restrict__ pooled, const float* __restrict__ counts,
                            const float* __restrict__ Wl, const float* __restrict__ bl,
                            float* __restrict__ out)
{
    int g = blockIdx.x * blockDim.x + threadIdx.x;
    if (g >= N_GRAPHS) return;
    float inv = 1.0f / fmaxf(counts[g], 1.0f);
    float logit[N_CLASSES];
#pragma unroll
    for (int k = 0; k < N_CLASSES; ++k) logit[k] = bl[k];
    for (int c = 0; c < HID; ++c) {
        float m = pooled[(size_t)g * HID + c] * inv;
#pragma unroll
        for (int k = 0; k < N_CLASSES; ++k) logit[k] += m * Wl[c * N_CLASSES + k];
    }
    float mx = logit[0];
#pragma unroll
    for (int k = 1; k < N_CLASSES; ++k) mx = fmaxf(mx, logit[k]);
    float s = 0.0f;
#pragma unroll
    for (int k = 0; k < N_CLASSES; ++k) { logit[k] = __expf(logit[k] - mx); s += logit[k]; }
    float is = 1.0f / s;
#pragma unroll
    for (int k = 0; k < N_CLASSES; ++k) out[(size_t)g * N_CLASSES + k] = logit[k] * is;
}

// ---------------- launcher ----------------
extern "C" void kernel_launch(void* const* d_in, const int* in_sizes, int n_in,
                              void* d_out, int out_size, void* d_ws, size_t ws_size,
                              hipStream_t stream) {
    const float* x     = (const float*)d_in[0];
    const int*   ei    = (const int*)  d_in[1];   // [2][E]: row0=src, row1=dst
    const int*   batch = (const int*)  d_in[2];
    const float* W1 = (const float*)d_in[3];  const float* b1 = (const float*)d_in[4];
    const float* W2 = (const float*)d_in[5];  const float* b2 = (const float*)d_in[6];
    const float* W3 = (const float*)d_in[7];  const float* b3 = (const float*)d_in[8];
    const float* Wl = (const float*)d_in[9];  const float* bl = (const float*)d_in[10];
    float* out = (float*)d_out;

    const int N = N_NODES, E = N_EDGES;

    // workspace layout (float offsets)
    float* ws      = (float*)d_ws;
    float* deg     = ws;                       // 50000 (becomes dis) -> pad 50048
    float* counts  = ws + 50048;               // 512
    float* pooled  = ws + 50560;               // 512*128 = 65536
    float* hbuf    = ws + 116096;              // 50000*128
    float* aggbuf  = hbuf + (size_t)N * HID;   // 50000*128
    unsigned short* wt = (unsigned short*)(aggbuf + (size_t)N * HID); // 128*128 bf16
    unsigned short* xb = wt + HID * HID;                              // 50000*128 bf16

    const int T = 256;

    // zero deg + counts + pooled (contiguous prefix)
    zero_f32<<<(116096 + T - 1) / T, T, 0, stream>>>(ws, 116096);

    // degree on dst (+1 self-loop in finish), then dis = rsqrt(deg+1)
    deg_count<<<(E + T - 1) / T, T, 0, stream>>>(ei + E, deg, E);
    finish_dis<<<(N + T - 1) / T, T, 0, stream>>>(deg, N);

    const float* Ws[3] = {W1, W2, W3};
    const float* bs[3] = {b1, b2, b3};
    const float* Fin = x;
    for (int layer = 0; layer < 3; ++layer) {
        int relu_in = (layer > 0) ? 1 : 0;   // ReLU applied to inputs of layers 2,3
        cvt_x<<<(N * HID + T - 1) / T, T, 0, stream>>>(Fin, xb, N * HID, relu_in);
        cvt_wt<<<(HID * HID + T - 1) / T, T, 0, stream>>>(Ws[layer], wt);
        gemm_bf16_wmma<<<N / 16, T, 0, stream>>>(xb, wt, hbuf, N);
        selfloop_init<<<(N * HID + T - 1) / T, T, 0, stream>>>(hbuf, deg, bs[layer], aggbuf, N * HID);
        scatter_edges<<<(E * 32) / T, T, 0, stream>>>(ei, ei + E, deg, hbuf, aggbuf, E);
        Fin = aggbuf;
    }

    // mean pool + head
    pool_nodes<<<(N * 32 + T - 1) / T, T, 0, stream>>>(aggbuf, batch, pooled, counts, N);
    head_kernel<<<(N_GRAPHS + T - 1) / T, T, 0, stream>>>(pooled, counts, Wl, bl, out);
}
